// MultiHeadAttentionBlock_49177375539658
// MI455X (gfx1250) — matrix operand order repaired and checked by reading, not compiled, not run on hardware
//
#include <hip/hip_runtime.h>
#include <hip/hip_bf16.h>
#include <stdint.h>

// ---------------------------------------------------------------------------
// MultiHeadAttentionBlock with clipped relative position bias (no softmax).
// All heavy GEMMs run through v_wmma_f32_16x16x32_bf16 (CDNA5 / gfx1250),
// double-buffered LDS pipeline; bf16 x bf16 GEMMs use async global->LDS
// copies (ASYNCcnt) when the toolchain exposes the gfx1250 builtins.
// ---------------------------------------------------------------------------

#define BATCH  2
#define SEQ    1024
#define DMODEL 1024
#define NHEAD  16
#define DHEAD  64
#define NREL   21      // 2*MAX_REL+1
#define MAXREL 10

typedef __attribute__((ext_vector_type(16))) __bf16 v16bf;
typedef __attribute__((ext_vector_type(8)))  float  v8f;
typedef __attribute__((ext_vector_type(8)))  unsigned short us8;

union FragBF {
    v16bf bf;
    us8   h[2];
};

__device__ __forceinline__ float bf2f(unsigned short u) {
    union { unsigned u; float f; } v;
    v.u = ((unsigned)u) << 16;
    return v.f;
}

// hardware f32->bf16 convert (RNE) instead of manual bit twiddling
__device__ __forceinline__ unsigned short f2bf(float f) {
    union { __bf16 h; unsigned short u; } v;
    v.h = (__bf16)f;
    return v.u;
}

// ---- optional CDNA5 async global->LDS path (guarded, sync fallback) ----
#if defined(__gfx1250__) && \
    __has_builtin(__builtin_amdgcn_global_load_async_to_lds_b128) && \
    __has_builtin(__builtin_amdgcn_s_wait_asynccnt)
#define HAVE_ASYNC 1
#else
#define HAVE_ASYNC 0
#endif

#if HAVE_ASYNC
// exact parameter types per hipcc diagnostic:
//   param0: int __vector(4) addrspace(1)* (global), param1: LDS counterpart
typedef int v4i_vs __attribute__((vector_size(16)));
typedef __attribute__((address_space(1))) v4i_vs* gv4i_t;
typedef __attribute__((address_space(3))) v4i_vs* lv4i_t;

__device__ __forceinline__ void async_cp_32B(const unsigned short* g, unsigned short* l) {
    gv4i_t ga = (gv4i_t)(void*)g;   // drop const, then generic -> AS1 v4i*
    lv4i_t la = (lv4i_t)(void*)l;   // generic -> AS3 v4i*
    __builtin_amdgcn_global_load_async_to_lds_b128(ga,     la,     0, 0);
    __builtin_amdgcn_global_load_async_to_lds_b128(ga + 1, la + 1, 0, 0);
}
#endif

// Epilogue modes
#define MODE_PROJ_HS 0   // bf16 store, layout [b,h,s,d]   (Q, K projections)
#define MODE_PROJ_T  1   // bf16 store, layout [b,h,d,s]   (V projection, transposed)
#define MODE_SCORES  2   // bf16 store attn = (QK^T + relP)/sqrt(DMODEL)
#define MODE_W1      3   // f32 store weight1 into xcat [b,s,h*64+d]
#define MODE_FINAL   4   // f32 store out + b0

// Tile geometry: 256 threads = 8 waves (4Mx2N), block tile 64x64, K-stage 64.
// Each wave: 16(M) x 32(N) -> 2 accumulators, 4 WMMA per stage.
#define BM  64
#define BN  64
#define BK  64
#define STR 72   // LDS row stride in ushort (144B: bank spread, 16B aligned)

// NT GEMM: C[m,n] = sum_k A[m,k]*B[n,k], fp32 accumulate via bf16 WMMA.
template <bool AF32, bool BF32, int MODE>
__global__ __launch_bounds__(256)
void gemm_nt(const void* __restrict__ Av, const void* __restrict__ Bv,
             const float* __restrict__ aux,       // bias (PROJ/FINAL) or P (SCORES)
             void* __restrict__ Outv,
             int K, int lda, int ldb,
             size_t strideAz, size_t strideBz)
{
    __shared__ __align__(16) unsigned short As[2][BM * STR];
    __shared__ __align__(16) unsigned short Bs[2][BN * STR];

    const int tid    = threadIdx.x;
    const int z      = blockIdx.z;
    const int blockM = blockIdx.y * BM;
    const int blockN = blockIdx.x * BN;

    const int wid  = tid >> 5;
    const int lane = tid & 31;
    const int g    = lane >> 4;        // lane group (0: lanes 0-15, 1: lanes 16-31)
    const int lm   = lane & 15;
    const int wm   = (wid >> 1) * 16;  // wave M offset within block (0/16/32/48)
    const int wn   = (wid & 1) * 32;   // wave N offset within block (0/32)

    // cooperative staging: 64 rows x 64 k, 16 contiguous elems per thread
    const int lr  = tid >> 2;          // 0..63
    const int lkc = (tid & 3) * 16;    // 0,16,32,48

    const size_t aRow = (size_t)z * strideAz + (size_t)(blockM + lr) * (size_t)lda;
    const size_t bRow = (size_t)z * strideBz + (size_t)(blockN + lr) * (size_t)ldb;

#if HAVE_ASYNC
    constexpr bool kAsync = (!AF32 && !BF32);
#else
    constexpr bool kAsync = false;
#endif

    // register staging for the sync pipeline
    float          fregA[16], fregB[16];
    unsigned short uregA[16], uregB[16];

    auto load_regs = [&](int k0) {
        #pragma unroll
        for (int j = 0; j < 16; ++j) {
            if constexpr (AF32) fregA[j] = ((const float*)Av)[aRow + (size_t)(k0 + lkc + j)];
            else                uregA[j] = ((const unsigned short*)Av)[aRow + (size_t)(k0 + lkc + j)];
        }
        #pragma unroll
        for (int j = 0; j < 16; ++j) {
            if constexpr (BF32) fregB[j] = ((const float*)Bv)[bRow + (size_t)(k0 + lkc + j)];
            else                uregB[j] = ((const unsigned short*)Bv)[bRow + (size_t)(k0 + lkc + j)];
        }
    };
    auto store_lds = [&](int buf) {
        #pragma unroll
        for (int j = 0; j < 16; ++j) {
            As[buf][lr * STR + lkc + j] = AF32 ? f2bf(fregA[j]) : uregA[j];
        }
        #pragma unroll
        for (int j = 0; j < 16; ++j) {
            Bs[buf][lr * STR + lkc + j] = BF32 ? f2bf(fregB[j]) : uregB[j];
        }
    };

    v8f acc0 = {};
    v8f acc1 = {};

    auto compute = [&](int cur) {
        #pragma unroll
        for (int kk = 0; kk < BK; kk += 32) {
            FragBF a, bb0, bb1;
            // A (16x32): elem j -> K = kk + (j/8)*16 + g*8 + (j%8)
            const unsigned short* as = &As[cur][(wm + lm) * STR + kk];
            a.h[0] = *(const us8*)(as + g * 8);
            a.h[1] = *(const us8*)(as + 16 + g * 8);
            // B (32x16): elem j -> K = kk + (j/8)*16 + g*8 + (j%8), column = lm
            const unsigned short* bs0 = &Bs[cur][(wn + lm) * STR + kk];
            bb0.h[0] = *(const us8*)(bs0 + g * 8);
            bb0.h[1] = *(const us8*)(bs0 + 16 + g * 8);
            const unsigned short* bs1 = &Bs[cur][(wn + 16 + lm) * STR + kk];
            bb1.h[0] = *(const us8*)(bs1 + g * 8);
            bb1.h[1] = *(const us8*)(bs1 + 16 + g * 8);

            acc0 = __builtin_amdgcn_wmma_f32_16x16x32_bf16(false, a.bf, false, bb0.bf,
                                                           (short)0, acc0, false, false);
            acc1 = __builtin_amdgcn_wmma_f32_16x16x32_bf16(false, a.bf, false, bb1.bf,
                                                           (short)0, acc1, false, false);
        }
    };

    // ---- prologue: stage 0 ----
#if HAVE_ASYNC
    if constexpr (kAsync) {
        async_cp_32B((const unsigned short*)Av + aRow + lkc, &As[0][lr * STR + lkc]);
        async_cp_32B((const unsigned short*)Bv + bRow + lkc, &Bs[0][lr * STR + lkc]);
        __builtin_amdgcn_s_wait_asynccnt(0);
    }
#endif
    if constexpr (!kAsync) {
        load_regs(0);
        store_lds(0);
    }
    __syncthreads();

    const int nk = K / BK;
    for (int s = 0; s < nk; ++s) {
        const int  cur  = s & 1;
        const bool more = (s + 1 < nk);

        // prefetch next stage while computing the current one
#if HAVE_ASYNC
        if constexpr (kAsync) {
            if (more) {
                const int k0 = (s + 1) * BK;
                async_cp_32B((const unsigned short*)Av + aRow + (size_t)(k0 + lkc),
                             &As[cur ^ 1][lr * STR + lkc]);
                async_cp_32B((const unsigned short*)Bv + bRow + (size_t)(k0 + lkc),
                             &Bs[cur ^ 1][lr * STR + lkc]);
            }
        }
#endif
        if constexpr (!kAsync) {
            if (more) load_regs((s + 1) * BK);
        }

        compute(cur);

#if HAVE_ASYNC
        if constexpr (kAsync) {
            if (more) __builtin_amdgcn_s_wait_asynccnt(0);
        }
#endif
        if constexpr (!kAsync) {
            if (more) store_lds(cur ^ 1);
        }
        __syncthreads();
    }

    // ---- epilogue: C/D layout: VGPR r, lanes 0-15 -> M=r, lanes 16-31 -> M=8+r ----
    #pragma unroll
    for (int r = 0; r < 8; ++r) {
        const int m = blockM + wm + g * 8 + r;
        #pragma unroll
        for (int sub = 0; sub < 2; ++sub) {
            const int n = blockN + wn + sub * 16 + lm;
            float v = (sub == 0) ? acc0[r] : acc1[r];

            if constexpr (MODE == MODE_PROJ_HS) {
                v += aux[n];
                const size_t addr =
                    (((size_t)(m >> 10) * NHEAD + (size_t)(n >> 6)) * SEQ + (size_t)(m & 1023)) * DHEAD
                    + (size_t)(n & 63);
                ((unsigned short*)Outv)[addr] = f2bf(v);
            } else if constexpr (MODE == MODE_PROJ_T) {
                v += aux[n];
                const size_t addr =
                    (((size_t)(m >> 10) * NHEAD + (size_t)(n >> 6)) * DHEAD + (size_t)(n & 63)) * SEQ
                    + (size_t)(m & 1023);
                ((unsigned short*)Outv)[addr] = f2bf(v);
            } else if constexpr (MODE == MODE_SCORES) {
                const int q = m;
                int d = n - q;
                d = d < -MAXREL ? -MAXREL : (d > MAXREL ? MAXREL : d);
                v = (v + aux[((size_t)z * SEQ + (size_t)q) * NREL + (size_t)(d + MAXREL)]) * 0.03125f;
                ((unsigned short*)Outv)[((size_t)z * SEQ + (size_t)q) * SEQ + (size_t)n] = f2bf(v);
            } else if constexpr (MODE == MODE_W1) {
                const size_t addr =
                    ((size_t)(z >> 4) * SEQ + (size_t)m) * DMODEL + (size_t)((z & 15) * DHEAD + n);
                ((float*)Outv)[addr] = v;
            } else { // MODE_FINAL
                ((float*)Outv)[(size_t)m * DMODEL + (size_t)n] = v + aux[n];
            }
        }
    }
}

// P[bh, q, r] = Q[bh,q,:] . rel_k_table[r,:]
__global__ void rel_p_kernel(const unsigned short* __restrict__ Qh,
                             const float* __restrict__ relk,
                             float* __restrict__ P)
{
    const int t = blockIdx.x * blockDim.x + threadIdx.x;
    const int total = BATCH * NHEAD * SEQ * NREL;
    if (t >= total) return;
    const int r   = t % NREL;
    const int bhq = t / NREL;
    const unsigned short* qrow = Qh + (size_t)bhq * DHEAD;
    const float* rk = relk + r * DHEAD;
    float s = 0.f;
    #pragma unroll 8
    for (int d = 0; d < DHEAD; ++d) s += bf2f(qrow[d]) * rk[d];
    P[t] = s;
}

// weight2: per-row bucket sums of attn over clipped relative distance,
// then bucket @ rel_v_table, accumulated into xcat.
__global__ void rel_v_kernel(const unsigned short* __restrict__ attn,
                             const float* __restrict__ relv,
                             float* __restrict__ xcat)
{
    const int t = blockIdx.x * blockDim.x + threadIdx.x;
    if (t >= BATCH * NHEAD * SEQ) return;
    const int bh = t >> 10;
    const int q  = t & 1023;
    const unsigned short* row = attn + ((size_t)bh * SEQ + (size_t)q) * SEQ;

    float bucket[NREL];
    #pragma unroll
    for (int r = 1; r < NREL - 1; ++r) {
        const int k = q + r - MAXREL;
        bucket[r] = (k >= 0 && k < SEQ) ? bf2f(row[k]) : 0.f;
    }
    float s0 = 0.f;
    for (int k = 0; k <= q - MAXREL; ++k) s0 += bf2f(row[k]);
    bucket[0] = s0;
    float s20 = 0.f;
    for (int k = q + MAXREL; k < SEQ; ++k) s20 += bf2f(row[k]);
    bucket[NREL - 1] = s20;

    float* xr = xcat + ((size_t)(bh >> 4) * SEQ + (size_t)q) * DMODEL + (size_t)(bh & 15) * DHEAD;
    for (int d = 0; d < DHEAD; ++d) {
        float s = 0.f;
        #pragma unroll
        for (int r = 0; r < NREL; ++r) s += bucket[r] * relv[r * DHEAD + d];
        xr[d] += s;
    }
}

extern "C" void kernel_launch(void* const* d_in, const int* in_sizes, int n_in,
                              void* d_out, int out_size, void* d_ws, size_t ws_size,
                              hipStream_t stream)
{
    (void)in_sizes; (void)n_in; (void)out_size; (void)ws_size;

    const float* q    = (const float*)d_in[0];
    const float* k    = (const float*)d_in[1];
    const float* v    = (const float*)d_in[2];
    // d_in[3] = mask: no-op in the reference
    const float* Wq   = (const float*)d_in[4];
    const float* bq   = (const float*)d_in[5];
    const float* Wk   = (const float*)d_in[6];
    const float* bk   = (const float*)d_in[7];
    const float* Wv   = (const float*)d_in[8];
    const float* bv   = (const float*)d_in[9];
    const float* W0   = (const float*)d_in[10];
    const float* b0   = (const float*)d_in[11];
    const float* relk = (const float*)d_in[12];
    const float* relv = (const float*)d_in[13];

    // workspace layout (bytes)
    char* ws = (char*)d_ws;
    const size_t BHS = (size_t)BATCH * NHEAD * SEQ;
    unsigned short* Qh   = (unsigned short*)(ws);                  // 4 MB
    unsigned short* Kh   = (unsigned short*)(ws + (4u << 20));     // 4 MB
    unsigned short* Vt   = (unsigned short*)(ws + (8u << 20));     // 4 MB
    unsigned short* attn = (unsigned short*)(ws + (12u << 20));    // 64 MB
    float*          P    = (float*)         (ws + (76u << 20));    // ~2.8 MB
    float*          xcat = (float*)         (ws + (80u << 20));    // 8 MB

    const dim3 blk(256);
    const size_t headStride = (size_t)SEQ * DHEAD;

    // 1-3) Q/K/V projections: [2048x1024] @ W^T, K=1024
    gemm_nt<true, true, MODE_PROJ_HS><<<dim3(16, 32, 1), blk, 0, stream>>>(
        q, Wq, bq, Qh, DMODEL, DMODEL, DMODEL, 0, 0);
    gemm_nt<true, true, MODE_PROJ_HS><<<dim3(16, 32, 1), blk, 0, stream>>>(
        k, Wk, bk, Kh, DMODEL, DMODEL, DMODEL, 0, 0);
    gemm_nt<true, true, MODE_PROJ_T><<<dim3(16, 32, 1), blk, 0, stream>>>(
        v, Wv, bv, Vt, DMODEL, DMODEL, DMODEL, 0, 0);

    // 4) relative-key dot table P[bh,q,r]
    {
        const int total = (int)(BHS * NREL);
        rel_p_kernel<<<(total + 255) / 256, 256, 0, stream>>>(Qh, relk, P);
    }

    // 5) attn = (Q K^T + P-gather) / sqrt(DMODEL), per (b,h), K=64
    gemm_nt<false, false, MODE_SCORES><<<dim3(16, 16, BATCH * NHEAD), blk, 0, stream>>>(
        Qh, Kh, P, attn, DHEAD, DHEAD, DHEAD, headStride, headStride);

    // 6) weight1 = attn @ V  (V pre-transposed -> NT, K=1024)
    gemm_nt<false, false, MODE_W1><<<dim3(1, 16, BATCH * NHEAD), blk, 0, stream>>>(
        attn, Vt, nullptr, xcat, SEQ, SEQ, SEQ,
        (size_t)SEQ * SEQ, (size_t)DHEAD * SEQ);

    // 7) weight2 accumulation into xcat
    rel_v_kernel<<<((int)BHS + 255) / 256, 256, 0, stream>>>(attn, relv, xcat);

    // 8) output projection: xcat @ W0^T + b0 -> d_out (fp32)
    gemm_nt<true, true, MODE_FINAL><<<dim3(16, 32, 1), blk, 0, stream>>>(
        xcat, W0, b0, d_out, DMODEL, DMODEL, DMODEL, 0, 0);
}
